// SinkhornMixer_85298050498979
// MI455X (gfx1250) — compile-verified, matrix operand order checked by reading, not run on hardware
//
#include <hip/hip_runtime.h>

#define N 8192
#define ITERS 5
#define EPS 1e-6f
#define RS_SLICES 8                 // split-K slices for row-sum pass
#define CS_SLICES 128               // row slices for col-sum pass
#define KSLICE (N / RS_SLICES)      // 1024
#define CS_ROWS (N / CS_SLICES)     // 64

typedef _Float16 v16h __attribute__((ext_vector_type(16)));
typedef _Float16 v8h  __attribute__((ext_vector_type(8)));
typedef float    v8f  __attribute__((ext_vector_type(8)));
typedef float    v4f  __attribute__((ext_vector_type(4)));

__device__ __forceinline__ float sigmoidf(float x) {
  return 1.0f / (1.0f + __expf(-x));
}

// ---------------------------------------------------------------------------
// init: r = c = 1, ch = 1 (fp16 copy of c used as WMMA B operand)
// ---------------------------------------------------------------------------
__global__ void init_vec_k(float* __restrict__ r, float* __restrict__ c,
                           _Float16* __restrict__ ch) {
  int i = blockIdx.x * blockDim.x + threadIdx.x;
  r[i] = 1.0f;
  c[i] = 1.0f;
  ch[i] = (_Float16)1.0f;
}

// ---------------------------------------------------------------------------
// convert: M0h = fp16( sigmoid(logits) + I ).  8 elements / thread,
// b128 in (x2, non-temporal: logits must not evict M0h from L2),
// b128 out (regular temporal: we WANT M0h resident in L2).
// ---------------------------------------------------------------------------
__global__ void convert_k(const float* __restrict__ logits,
                          _Float16* __restrict__ m0) {
  size_t idx = ((size_t)blockIdx.x * blockDim.x + threadIdx.x) * 8;
  int row = (int)(idx >> 13);        // / N
  int col = (int)(idx & (N - 1));
  v4f x0 = __builtin_nontemporal_load((const v4f*)(logits + idx));
  v4f x1 = __builtin_nontemporal_load((const v4f*)(logits + idx + 4));
  v8h o;
#pragma unroll
  for (int t = 0; t < 4; ++t)
    o[t] = (_Float16)(sigmoidf(x0[t]) + ((col + t == row) ? 1.0f : 0.0f));
#pragma unroll
  for (int t = 0; t < 4; ++t)
    o[4 + t] = (_Float16)(sigmoidf(x1[t]) + ((col + 4 + t == row) ? 1.0f : 0.0f));
  *(v8h*)(m0 + idx) = o;
}

// ---------------------------------------------------------------------------
// row-sum matvec via WMMA:  ypart[slice][i] = sum_{k in slice} M0h[i][k]*c[k]
// One wave handles a 16-row tile, K-loop of 32 per v_wmma_f32_16x16x32_f16.
// A layout (ISA 16-bit A 16x32): lane L(&15) = row, bytes (lane>=16?16:0)+{0,32}
// within the 64-byte K-chunk.  B (32x16, c replicated over all 16 columns)
// has the *same* per-lane byte pattern over a 64-byte chunk of ch.
// ---------------------------------------------------------------------------
__global__ void rowsum_wmma_k(const _Float16* __restrict__ m0,
                              const _Float16* __restrict__ ch,
                              float* __restrict__ ypart) {
  const int lane    = threadIdx.x & 31;
  const int wave    = threadIdx.x >> 5;
  const int tile    = blockIdx.x * 8 + wave;     // 512 tiles of 16 rows
  const int rowbase = tile * 16;
  const int row     = rowbase + (lane & 15);
  const int koff    = (lane >> 4) * 16;          // byte offset within chunk
  const size_t kbyte = (size_t)blockIdx.y * KSLICE * 2;

  const char* aptr = (const char*)(m0 + (size_t)row * N) + kbyte + koff;
  const char* bptr = (const char*)ch + kbyte + koff;

  v8f acc = {};
#pragma unroll 4
  for (int k = 0; k < KSLICE; k += 32) {
    v8h a0 = *(const v8h*)(aptr);
    v8h a1 = *(const v8h*)(aptr + 32);
    v8h b0 = *(const v8h*)(bptr);
    v8h b1 = *(const v8h*)(bptr + 32);
    v16h A = __builtin_shufflevector(a0, a1, 0, 1, 2, 3, 4, 5, 6, 7,
                                     8, 9, 10, 11, 12, 13, 14, 15);
    v16h B = __builtin_shufflevector(b0, b1, 0, 1, 2, 3, 4, 5, 6, 7,
                                     8, 9, 10, 11, 12, 13, 14, 15);
    acc = __builtin_amdgcn_wmma_f32_16x16x32_f16(false, A, false, B,
                                                 (short)0, acc, false, false);
    // high-locality prefetch: pull upcoming A bytes from L2 into WGP cache
    __builtin_prefetch(aptr + 1024, 0, 3);
    aptr += 64;
    bptr += 64;
  }
  // D replicated across N: lane 0 holds rows 0..7 (VGPR v -> M=v),
  // lane 16 holds rows 8..15.
  if ((lane & 15) == 0) {
    float* dst = ypart + (size_t)blockIdx.y * N + rowbase + (lane >> 4) * 8;
    v4f lo = __builtin_shufflevector(acc, acc, 0, 1, 2, 3);
    v4f hi = __builtin_shufflevector(acc, acc, 4, 5, 6, 7);
    *(v4f*)dst = lo;
    *((v4f*)dst + 1) = hi;
  }
}

// ---------------------------------------------------------------------------
// col-sum matvec:  zpart[slice][j] = sum_{i in slice} M0h[i][j] * r[i]
// Fully coalesced: thread owns 8 consecutive columns (one b128 of fp16),
// r[i] is wave-uniform (scalar load).
// ---------------------------------------------------------------------------
__global__ void colsum_h_k(const _Float16* __restrict__ m0,
                           const float* __restrict__ r,
                           float* __restrict__ zpart) {
  const int col8    = (blockIdx.x * 256 + threadIdx.x) * 8;
  const int rowbase = blockIdx.y * CS_ROWS;
  float acc[8] = {0.f, 0.f, 0.f, 0.f, 0.f, 0.f, 0.f, 0.f};
  for (int i = 0; i < CS_ROWS; ++i) {
    float rv = r[rowbase + i];  // uniform -> SGPR
    v8h m = *(const v8h*)(m0 + (size_t)(rowbase + i) * N + col8);
#pragma unroll
    for (int t = 0; t < 8; ++t) acc[t] += (float)m[t] * rv;
  }
  float* dst = zpart + (size_t)blockIdx.y * N + col8;
  v4f lo = {acc[0], acc[1], acc[2], acc[3]};
  v4f hi = {acc[4], acc[5], acc[6], acc[7]};
  *(v4f*)dst = lo;
  *((v4f*)dst + 1) = hi;
}

// ---------------------------------------------------------------------------
// r update:  r_i = r_i / (r_i * sum_s ypart[s][i] + eps)
// ---------------------------------------------------------------------------
__global__ void update_r_k(const float* __restrict__ ypart,
                           float* __restrict__ r, int ns) {
  int i = blockIdx.x * blockDim.x + threadIdx.x;
  float s = 0.f;
  for (int q = 0; q < ns; ++q) s += ypart[(size_t)q * N + i];
  float rv = r[i];
  r[i] = rv / (rv * s + EPS);
}

// ---------------------------------------------------------------------------
// c update:  c_j = c_j / (c_j * sum_s zpart[s][j] + eps);  ch = fp16(c)
// ---------------------------------------------------------------------------
__global__ void update_c_k(const float* __restrict__ zpart,
                           float* __restrict__ c, _Float16* __restrict__ ch,
                           int ns) {
  int j = blockIdx.x * blockDim.x + threadIdx.x;
  float s = 0.f;
  for (int q = 0; q < ns; ++q) s += zpart[(size_t)q * N + j];
  float cv = c[j];
  float cn = cv / (cv * s + EPS);
  c[j] = cn;
  ch[j] = (_Float16)cn;
}

// ---------------------------------------------------------------------------
// final:  out[i][j] = (sigmoid(logits[i][j]) + I) * r_i * c_j   (full f32)
// Non-temporal in and out: pure stream, never reused.
// ---------------------------------------------------------------------------
__global__ void final_k(const float* __restrict__ logits,
                        const float* __restrict__ r,
                        const float* __restrict__ c,
                        float* __restrict__ out) {
  size_t idx = ((size_t)blockIdx.x * blockDim.x + threadIdx.x) * 4;
  int row = (int)(idx >> 13);
  int col = (int)(idx & (N - 1));
  float rv = r[row];
  v4f x = __builtin_nontemporal_load((const v4f*)(logits + idx));
  v4f cv = *(const v4f*)(c + col);
  v4f o;
#pragma unroll
  for (int t = 0; t < 4; ++t) {
    float s = sigmoidf(x[t]) + ((col + t == row) ? 1.0f : 0.0f);
    o[t] = s * rv * cv[t];
  }
  __builtin_nontemporal_store(o, (v4f*)(out + idx));
}

// ---------------------------------------------------------------------------
// Fallback path (workspace too small for fp16 M0): recompute sigmoid on the
// fly from f32 logits each pass.
// ---------------------------------------------------------------------------
__global__ void rowsum_f32_k(const float* __restrict__ logits,
                             const float* __restrict__ c,
                             float* __restrict__ ypart) {
  const int lane = threadIdx.x & 31;
  const int wave = threadIdx.x >> 5;
  const int row  = blockIdx.x * 8 + wave;
  const float* lr = logits + (size_t)row * N;
  float acc = 0.f;
  for (int j = lane * 4; j < N; j += 128) {
    v4f x = *(const v4f*)(lr + j);
    v4f cv = *(const v4f*)(c + j);
#pragma unroll
    for (int t = 0; t < 4; ++t) {
      float s = sigmoidf(x[t]) + ((j + t == row) ? 1.0f : 0.0f);
      acc += s * cv[t];
    }
  }
  for (int off = 16; off > 0; off >>= 1) acc += __shfl_down(acc, off);
  if (lane == 0) ypart[row] = acc;
}

__global__ void colsum_f32_k(const float* __restrict__ logits,
                             const float* __restrict__ r,
                             float* __restrict__ zpart) {
  const int col4    = (blockIdx.x * 256 + threadIdx.x) * 4;
  const int rowbase = blockIdx.y * CS_ROWS;
  float acc[4] = {0.f, 0.f, 0.f, 0.f};
  for (int i = 0; i < CS_ROWS; ++i) {
    int row = rowbase + i;
    float rv = r[row];  // uniform -> SGPR
    v4f x = *(const v4f*)(logits + (size_t)row * N + col4);
#pragma unroll
    for (int t = 0; t < 4; ++t) {
      float s = sigmoidf(x[t]) + ((col4 + t == row) ? 1.0f : 0.0f);
      acc[t] += s * rv;
    }
  }
  v4f o = {acc[0], acc[1], acc[2], acc[3]};
  *(v4f*)(zpart + (size_t)blockIdx.y * N + col4) = o;
}

// ---------------------------------------------------------------------------
extern "C" void kernel_launch(void* const* d_in, const int* in_sizes, int n_in,
                              void* d_out, int out_size, void* d_ws, size_t ws_size,
                              hipStream_t stream) {
  const float* logits = (const float*)d_in[0];
  float* out = (float*)d_out;
  char* ws = (char*)d_ws;

  const size_t M0_BYTES = (size_t)N * N * sizeof(_Float16);   // 128 MB
  const size_t VEC_BYTES = (size_t)N * 4          // r
                         + (size_t)N * 4          // c
                         + (size_t)N * 2          // ch
                         + (size_t)RS_SLICES * N * 4   // ypart
                         + (size_t)CS_SLICES * N * 4;  // zpart
  const bool big = ws_size >= M0_BYTES + VEC_BYTES;
  const size_t base = big ? M0_BYTES : 0;

  _Float16* m0   = (_Float16*)ws;
  float*    r    = (float*)(ws + base);
  float*    c    = r + N;
  _Float16* ch   = (_Float16*)(c + N);
  float*    ypart = (float*)((char*)ch + (size_t)N * 2);
  float*    zpart = ypart + (size_t)RS_SLICES * N;

  init_vec_k<<<N / 256, 256, 0, stream>>>(r, c, ch);
  if (big) {
    convert_k<<<(int)(((size_t)N * N / 8) / 256), 256, 0, stream>>>(logits, m0);
  }

  for (int it = 0; it < ITERS; ++it) {
    if (big) {
      rowsum_wmma_k<<<dim3(N / (16 * 8), RS_SLICES), 256, 0, stream>>>(m0, ch, ypart);
    } else {
      rowsum_f32_k<<<N / 8, 256, 0, stream>>>(logits, c, ypart);
    }
    update_r_k<<<N / 256, 256, 0, stream>>>(ypart, r, big ? RS_SLICES : 1);

    if (big) {
      colsum_h_k<<<dim3(N / (256 * 8), CS_SLICES), 256, 0, stream>>>(m0, r, zpart);
    } else {
      colsum_f32_k<<<dim3(N / (256 * 4), CS_SLICES), 256, 0, stream>>>(logits, r, zpart);
    }
    update_c_k<<<N / 256, 256, 0, stream>>>(zpart, c, ch, CS_SLICES);
  }

  final_k<<<(int)(((size_t)N * N / 4) / 256), 256, 0, stream>>>(logits, r, c, out);
}
